// HGCLEncoder_55997783605349
// MI455X (gfx1250) — matrix operand order repaired and checked by reading, not compiled, not run on hardware
//
#include <hip/hip_runtime.h>

// ---------------------------------------------------------------------------
// Types for CDNA5 WMMA (wave32): 16x16x32 bf16 -> f32
// ---------------------------------------------------------------------------
typedef __bf16 v16bf __attribute__((ext_vector_type(16)));
typedef float  v8f   __attribute__((ext_vector_type(8)));

typedef unsigned int u32x4 __attribute__((ext_vector_type(4)));
typedef int          i32x8 __attribute__((ext_vector_type(8)));
typedef int          i32x4 __attribute__((ext_vector_type(4)));

union Frag { v16bf v; uint4 q[2]; };

__device__ __forceinline__ unsigned short f32_to_bf16(float f) {
    union { float f; unsigned int u; } c; c.f = f;
    unsigned int u = c.u;
    u += 0x7FFFu + ((u >> 16) & 1u);   // round to nearest even
    return (unsigned short)(u >> 16);
}

#if defined(__has_builtin)
#if __has_builtin(__builtin_amdgcn_tensor_load_to_lds) && __has_builtin(__builtin_amdgcn_s_wait_tensorcnt)
#define USE_TDM 1
#endif
#endif

#ifdef USE_TDM
__device__ __forceinline__ unsigned lds_off_u32(const void* p) {
    // generic -> LDS(addrspace 3) -> 32-bit LDS byte offset
    return (unsigned)(unsigned long long)(const __attribute__((address_space(3))) char*)p;
}

// 2D TDM tile load: tensor is [d1 rows][tensor stride0 elems] of 2-byte data,
// tile is tile_d1 x tile_d0 (row-major, contiguous into LDS).
// D# layout per CDNA5 ISA 8.3/8.4 (group0 128b, group1 256b).
__device__ __forceinline__ void tdm_load_2d(unsigned lds_off,
                                            const unsigned short* gaddr,
                                            unsigned tensor_d0, unsigned tensor_d1,
                                            unsigned tile_d0, unsigned tile_d1,
                                            unsigned stride0)
{
    unsigned long long ga = (unsigned long long)gaddr;
    u32x4 g0;
    g0[0] = 1u;                                          // count=1 (valid), no gather
    g0[1] = lds_off;                                     // lds_addr [63:32]
    g0[2] = (unsigned)(ga & 0xFFFFFFFFu);                // global_addr lo
    g0[3] = (unsigned)((ga >> 32) & 0x01FFFFFFu)         // global_addr [56:32]
          | (2u << 30);                                  // type=2 ("image")
    i32x8 g1;
    g1[0] = (int)(1u << 16);                             // wg_mask=0, data_size=1 (2B)
    g1[1] = (int)((tensor_d0 & 0xFFFFu) << 16);          // tensor_dim0 lo16 -> [63:48]
    g1[2] = (int)(((tensor_d0 >> 16) & 0xFFFFu)          // tensor_dim0 hi16 -> [79:64]
          | ((tensor_d1 & 0xFFFFu) << 16));              // tensor_dim1 lo16 -> [95:80]
    g1[3] = (int)(((tensor_d1 >> 16) & 0xFFFFu)          // tensor_dim1 hi16 -> [111:96]
          | ((tile_d0 & 0xFFFFu) << 16));                // tile_dim0 -> [127:112]
    g1[4] = (int)(tile_d1 & 0xFFFFu);                    // tile_dim1 -> [143:128], tile_dim2=0
    g1[5] = (int)stride0;                                // tensor_dim0_stride lo32
    g1[6] = 0;                                           // stride hi16, dim1_stride lo16
    g1[7] = 0;
    i32x4 z4 = { 0, 0, 0, 0 };
    i32x8 z8 = { 0, 0, 0, 0, 0, 0, 0, 0 };
    // 6-arg form (clang-23 / therock-10.0 HIP headers)
    __builtin_amdgcn_tensor_load_to_lds(g0, g1, z4, z4, z8, 0);
}
#endif

// ---------------------------------------------------------------------------
// WMMA GEMM: C[M x 128] = Abf[M x K](bf16) * Bt[128 x K](bf16, n-major)
//            (+ bias, optional relu)
// 256 threads = 8 waves; block computes a 128-row x 128-col tile.
// A tile and B tile are both staged in LDS as [128][32] bf16 row-major so
// WMMA fragments are contiguous 16B/32B LDS reads (ISA 7.12.2 layouts).
// ---------------------------------------------------------------------------
template<int RELU>
__global__ __launch_bounds__(256)
void wmma_gemm128(const unsigned short* __restrict__ Abf,
                  const unsigned short* __restrict__ Bt,
                  const float* __restrict__ bias, float* __restrict__ C,
                  int M, int K)
{
    __shared__ unsigned short As[128 * 32];   // A tile: row r, K chunk of 32
    __shared__ unsigned short Bs[128 * 32];   // B tile: col n, K chunk of 32

    const int tid  = threadIdx.x;
    const int lane = tid & 31;
    const int wave = tid >> 5;
    const int half = lane >> 4;
    const int ml   = lane & 15;
    const int rowBase = blockIdx.x * 128;
#ifdef USE_TDM
    const unsigned Mrem = (unsigned)(M - rowBase);   // >= 1
#endif

    v8f zero = {};
    v8f acc[8];
#pragma unroll
    for (int t = 0; t < 8; ++t) acc[t] = zero;

    for (int k0 = 0; k0 < K; k0 += 32) {
#ifdef USE_TDM
        // Tensor Data Mover: 2D tile loads straight into LDS.
        // tensor_dim1 = remaining rows => OOB rows read as zero (ragged tail).
        if (wave == 0)
            tdm_load_2d(lds_off_u32(As), Abf + (size_t)rowBase * K + k0,
                        (unsigned)(K - k0), Mrem, 32u, 128u, (unsigned)K);
        if (wave == 1)
            tdm_load_2d(lds_off_u32(Bs), Bt + k0,
                        (unsigned)(K - k0), 128u, 32u, 128u, (unsigned)K);
        if (wave < 2)
            __builtin_amdgcn_s_wait_tensorcnt(0);
#else
        // synchronous bf16 tile copy fallback (uint4 = 8 bf16 per thread)
#pragma unroll
        for (int it = 0; it < 2; ++it) {
            int l8 = (it * 256 + tid) * 8;
            int r = l8 >> 5, c = l8 & 31;
            int gr = rowBase + r;
            uint4 v = make_uint4(0u, 0u, 0u, 0u);
            if (gr < M) v = *(const uint4*)(Abf + (size_t)gr * K + k0 + c);
            *(uint4*)&As[r * 32 + c] = v;
        }
#pragma unroll
        for (int it = 0; it < 2; ++it) {
            int l8 = (it * 256 + tid) * 8;
            int r = l8 >> 5, c = l8 & 31;
            uint4 v = *(const uint4*)(Bt + (size_t)r * K + k0 + c);
            *(uint4*)&Bs[r * 32 + c] = v;
        }
#endif
        __syncthreads();

        // A fragment (16x32 bf16): regs 0-3 = K[8h..8h+7], regs 4-7 = K[16+8h..+7]
        Frag a;
        const unsigned short* ap = &As[(16 * wave + ml) * 32 + 8 * half];
        a.q[0] = *(const uint4*)(ap);
        a.q[1] = *(const uint4*)(ap + 16);

#pragma unroll
        for (int t = 0; t < 8; ++t) {
            // B fragment: regs 0-3 = K[8h..8h+7], regs 4-7 = K[16+8h..+7]
            // at column n = 16t+ml, in the K-major tile.
            Frag b;
            const unsigned short* bp = &Bs[(16 * t + ml) * 32 + 8 * half];
            b.q[0] = *(const uint4*)(bp);
            b.q[1] = *(const uint4*)(bp + 16);
            acc[t] = __builtin_amdgcn_wmma_f32_16x16x32_bf16(
                false, a.v, false, b.v, (short)0, acc[t], false, false);
        }
        __syncthreads();
    }

    // C layout (ISA 7.12.2): reg v -> row v + 8*half, col = ml
#pragma unroll
    for (int t = 0; t < 8; ++t) {
        int col = 16 * t + ml;
        float bv = bias ? bias[col] : 0.f;
#pragma unroll
        for (int r = 0; r < 8; ++r) {
            int grow = rowBase + 16 * wave + r + 8 * half;
            if (grow < M) {
                float v = acc[t][r] + bv;
                if (RELU) v = fmaxf(v, 0.f);
                C[(size_t)grow * 128 + col] = v;
            }
        }
    }
}

// ---------------------------------------------------------------------------
// bf16 staging kernels
// ---------------------------------------------------------------------------
__global__ void f32_to_bf16_k(const float* __restrict__ s,
                              unsigned short* __restrict__ d, long long n4)
{
    long long i = (long long)blockIdx.x * blockDim.x + threadIdx.x;
    if (i >= n4) return;
    float4 v = *(const float4*)(s + i * 4);
    union { unsigned short u[4]; uint2 q; } p;
    p.u[0] = f32_to_bf16(v.x); p.u[1] = f32_to_bf16(v.y);
    p.u[2] = f32_to_bf16(v.z); p.u[3] = f32_to_bf16(v.w);
    *(uint2*)(d + i * 4) = p.q;
}

// B[K][128] f32 -> Bt[128][K] bf16
__global__ void transpose_bf16_k(const float* __restrict__ B,
                                 unsigned short* __restrict__ Bt, int K)
{
    int idx = blockIdx.x * blockDim.x + threadIdx.x;
    if (idx >= K * 128) return;
    int k = idx >> 7, n = idx & 127;
    Bt[(size_t)n * K + k] = f32_to_bf16(B[(size_t)k * 128 + n]);
}

// ---------------------------------------------------------------------------
// Elementwise / scatter helpers
// ---------------------------------------------------------------------------
__global__ void fill_k(float* __restrict__ p, float v, long long n) {
    long long i = (long long)blockIdx.x * blockDim.x + threadIdx.x;
    if (i < n) p[i] = v;
}

__global__ void count_k(const int* __restrict__ idx, long long n, float* __restrict__ deg) {
    long long i = (long long)blockIdx.x * blockDim.x + threadIdx.x;
    if (i < n) atomicAdd(&deg[idx[i]], 1.0f);
}

__global__ void rsqrt_k(float* __restrict__ d, long long n) {
    long long i = (long long)blockIdx.x * blockDim.x + threadIdx.x;
    if (i < n) d[i] = rsqrtf(d[i]);
}

__global__ void inv_or_zero_k(float* __restrict__ d, long long n) {
    long long i = (long long)blockIdx.x * blockDim.x + threadIdx.x;
    if (i < n) { float v = d[i]; d[i] = (v > 0.f) ? (1.0f / v) : 0.f; }
}

// out[i][c] = h[i][c] * dinv[i]^2 + bias[c]   (GCN self-loop + bias)
__global__ __launch_bounds__(256)
void gcn_self_init_k(const float* __restrict__ h, const float* __restrict__ dinv,
                     const float* __restrict__ bias, float* __restrict__ out,
                     long long n)
{
    long long g = (long long)blockIdx.x * blockDim.x + threadIdx.x;
    if (g >= n * 32) return;
    long long i = g >> 5; int c = (int)(g & 31) * 4;
    float di = dinv[i]; float w = di * di;
    float4 v  = *(const float4*)(h + i * 128 + c);
    float4 bv = *(const float4*)(bias + c);
    float4 o;
    o.x = v.x * w + bv.x; o.y = v.y * w + bv.y;
    o.z = v.z * w + bv.z; o.w = v.w * w + bv.w;
    *(float4*)(out + i * 128 + c) = o;
}

// out[dst[e]] += h[src[e]] * dinv[src]*dinv[dst]   (GCN edges)
__global__ __launch_bounds__(256)
void gcn_scatter_k(const float* __restrict__ h, const int* __restrict__ src,
                   const int* __restrict__ dst, const float* __restrict__ dinv,
                   float* __restrict__ out, long long E)
{
    long long g = (long long)blockIdx.x * blockDim.x + threadIdx.x;
    if (g >= E * 32) return;
    long long e = g >> 5; int c = (int)(g & 31) * 4;
    int s = src[e], d = dst[e];
    if ((g & 31) == 0 && e + 64 < E)
        __builtin_prefetch(h + (size_t)src[e + 64] * 128, 0, 1);  // global_prefetch_b8
    float nm = dinv[s] * dinv[d];
    float4 v = *(const float4*)(h + (size_t)s * 128 + c);
    float* o = out + (size_t)d * 128 + c;
    atomicAdd(o + 0, v.x * nm);
    atomicAdd(o + 1, v.y * nm);
    atomicAdd(o + 2, v.z * nm);
    atomicAdd(o + 3, v.w * nm);
}

// out[di[e]] += h[si[e]]   (hypergraph incidence scatter, both directions)
__global__ __launch_bounds__(256)
void scatter_rows_k(const float* __restrict__ h, const int* __restrict__ si,
                    const int* __restrict__ di, float* __restrict__ out, long long E)
{
    long long g = (long long)blockIdx.x * blockDim.x + threadIdx.x;
    if (g >= E * 32) return;
    long long e = g >> 5; int c = (int)(g & 31) * 4;
    int s = si[e], d = di[e];
    if ((g & 31) == 0 && e + 64 < E)
        __builtin_prefetch(h + (size_t)si[e + 64] * 128, 0, 1);
    float4 v = *(const float4*)(h + (size_t)s * 128 + c);
    float* o = out + (size_t)d * 128 + c;
    atomicAdd(o + 0, v.x);
    atomicAdd(o + 1, v.y);
    atomicAdd(o + 2, v.z);
    atomicAdd(o + 3, v.w);
}

__global__ __launch_bounds__(256)
void scale_rows_k(float* __restrict__ m, const float* __restrict__ sinv, long long rows)
{
    long long g = (long long)blockIdx.x * blockDim.x + threadIdx.x;
    if (g >= rows * 32) return;
    long long i = g >> 5; int c = (int)(g & 31) * 4;
    float s = sinv[i];
    float4 v = *(float4*)(m + i * 128 + c);
    v.x *= s; v.y *= s; v.z *= s; v.w *= s;
    *(float4*)(m + i * 128 + c) = v;
}

__global__ __launch_bounds__(256)
void scale_rows_bias_k(float* __restrict__ o, const float* __restrict__ sinv,
                       const float* __restrict__ bias, long long rows)
{
    long long g = (long long)blockIdx.x * blockDim.x + threadIdx.x;
    if (g >= rows * 32) return;
    long long i = g >> 5; int c = (int)(g & 31) * 4;
    float s = sinv[i];
    float4 v  = *(float4*)(o + i * 128 + c);
    float4 bv = *(const float4*)(bias + c);
    v.x = v.x * s + bv.x; v.y = v.y * s + bv.y;
    v.z = v.z * s + bv.z; v.w = v.w * s + bv.w;
    *(float4*)(o + i * 128 + c) = v;
}

// LayerNorm(128) + ReLU, one row per wave (wave32)
__global__ __launch_bounds__(256)
void ln_relu_k(float* __restrict__ h, const float* __restrict__ w,
               const float* __restrict__ b, long long n)
{
    int wave = threadIdx.x >> 5, lane = threadIdx.x & 31;
    long long row = (long long)blockIdx.x * 8 + wave;
    if (row >= n) return;
    float4 v = *(const float4*)(h + row * 128 + lane * 4);
    float s = v.x + v.y + v.z + v.w;
#pragma unroll
    for (int o = 16; o > 0; o >>= 1) s += __shfl_xor(s, o, 32);
    float mu = s * (1.0f / 128.0f);
    float dx = v.x - mu, dy = v.y - mu, dz = v.z - mu, dw = v.w - mu;
    float q = dx * dx + dy * dy + dz * dz + dw * dw;
#pragma unroll
    for (int o = 16; o > 0; o >>= 1) q += __shfl_xor(q, o, 32);
    float rs = rsqrtf(q * (1.0f / 128.0f) + 1e-5f);
    float4 wv = *(const float4*)(w + lane * 4);
    float4 bv = *(const float4*)(b + lane * 4);
    float4 o4;
    o4.x = fmaxf(dx * rs * wv.x + bv.x, 0.f);
    o4.y = fmaxf(dy * rs * wv.y + bv.y, 0.f);
    o4.z = fmaxf(dz * rs * wv.z + bv.z, 0.f);
    o4.w = fmaxf(dw * rs * wv.w + bv.w, 0.f);
    *(float4*)(h + row * 128 + lane * 4) = o4;
}

__global__ void avg_k(const float* __restrict__ a, const float* __restrict__ b,
                      float* __restrict__ o, long long n4)
{
    long long i = (long long)blockIdx.x * blockDim.x + threadIdx.x;
    if (i >= n4) return;
    float4 va = *(const float4*)(a + i * 4);
    float4 vb = *(const float4*)(b + i * 4);
    float4 vo;
    vo.x = 0.5f * (va.x + vb.x); vo.y = 0.5f * (va.y + vb.y);
    vo.z = 0.5f * (va.z + vb.z); vo.w = 0.5f * (va.w + vb.w);
    *(float4*)(o + i * 4) = vo;
}

// ---------------------------------------------------------------------------
// Host-side orchestration
// ---------------------------------------------------------------------------
static inline unsigned int cdiv_ll(long long a, long long b) {
    return (unsigned int)((a + b - 1) / b);
}

extern "C" void kernel_launch(void* const* d_in, const int* in_sizes, int n_in,
                              void* d_out, int out_size, void* d_ws, size_t ws_size,
                              hipStream_t stream) {
    (void)n_in; (void)out_size; (void)ws_size;
    const float* x    = (const float*)d_in[0];
    const int*   ei   = (const int*)d_in[1];
    const int*   hei  = (const int*)d_in[2];
    const float* W_g1 = (const float*)d_in[3];
    const float* b_g1 = (const float*)d_in[4];
    const float* W_g2 = (const float*)d_in[5];
    const float* b_g2 = (const float*)d_in[6];
    const float* W_h1 = (const float*)d_in[7];
    const float* b_h1 = (const float*)d_in[8];
    const float* W_h2 = (const float*)d_in[9];
    const float* b_h2 = (const float*)d_in[10];
    const float* ln1w = (const float*)d_in[11];
    const float* ln1b = (const float*)d_in[12];
    const float* ln2w = (const float*)d_in[13];
    const float* ln2b = (const float*)d_in[14];
    const float* Wp1  = (const float*)d_in[15];
    const float* bp1  = (const float*)d_in[16];
    const float* Wp2  = (const float*)d_in[17];
    const float* bp2  = (const float*)d_in[18];

    const long long N   = in_sizes[0] / 512;   // nodes
    const long long E   = in_sizes[1] / 2;     // graph edges
    const long long HM  = in_sizes[2] / 2;     // hypergraph incidences
    const long long NHE = 100000;              // hyperedge segment count (ref constant)

    const int* e_src   = ei;
    const int* e_dst   = ei + E;
    const int* h_node  = hei;
    const int* h_hedge = hei + HM;

    // workspace carve-up (256B aligned)
    size_t off = 0;
    auto carve = [&](size_t bytes) -> void* {
        void* p = (char*)d_ws + off;
        off += (bytes + 255) & ~(size_t)255;
        return p;
    };
    float* bufA = (float*)carve((size_t)N * 128 * 4);
    float* bufB = (float*)carve((size_t)N * 128 * 4);
    float* bufC = (float*)carve((size_t)N * 128 * 4);
    float* bufM = (float*)carve((size_t)NHE * 128 * 4);
    unsigned short* Abf = (unsigned short*)carve((size_t)N * 512 * 2);
    unsigned short* Btw = (unsigned short*)carve((size_t)128 * 512 * 2);
    float* dinv = (float*)carve((size_t)N * 4);
    float* Dinv = (float*)carve((size_t)N * 4);
    float* Binv = (float*)carve((size_t)NHE * 4);

    const dim3 blk(256);
    const unsigned int gemmBlocks = cdiv_ll(N, 128);

    // A(f32) @ B(f32) -> C, staging A,B as bf16 (TDM-friendly layouts)
    auto run_gemm = [&](const float* A, const float* B, const float* bias,
                        float* C, int K, int relu) {
        f32_to_bf16_k<<<cdiv_ll(N * K / 4, 256), blk, 0, stream>>>(A, Abf, N * K / 4);
        transpose_bf16_k<<<cdiv_ll((long long)K * 128, 256), blk, 0, stream>>>(B, Btw, K);
        if (relu)
            wmma_gemm128<1><<<gemmBlocks, blk, 0, stream>>>(Abf, Btw, bias, C, (int)N, K);
        else
            wmma_gemm128<0><<<gemmBlocks, blk, 0, stream>>>(Abf, Btw, bias, C, (int)N, K);
    };

    // ---- normalizers
    fill_k<<<cdiv_ll(N, 256), blk, 0, stream>>>(dinv, 1.0f, N);       // self loop
    count_k<<<cdiv_ll(E, 256), blk, 0, stream>>>(e_dst, E, dinv);
    rsqrt_k<<<cdiv_ll(N, 256), blk, 0, stream>>>(dinv, N);

    fill_k<<<cdiv_ll(N, 256), blk, 0, stream>>>(Dinv, 0.0f, N);
    count_k<<<cdiv_ll(HM, 256), blk, 0, stream>>>(h_node, HM, Dinv);
    inv_or_zero_k<<<cdiv_ll(N, 256), blk, 0, stream>>>(Dinv, N);

    fill_k<<<cdiv_ll(NHE, 256), blk, 0, stream>>>(Binv, 0.0f, NHE);
    count_k<<<cdiv_ll(HM, 256), blk, 0, stream>>>(h_hedge, HM, Binv);
    inv_or_zero_k<<<cdiv_ll(NHE, 256), blk, 0, stream>>>(Binv, NHE);

    // ---- GCN branch: layer 1
    run_gemm(x, W_g1, nullptr, bufA, 512, 0);
    gcn_self_init_k<<<cdiv_ll(N * 32, 256), blk, 0, stream>>>(bufA, dinv, b_g1, bufB, N);
    gcn_scatter_k<<<cdiv_ll(E * 32, 256), blk, 0, stream>>>(bufA, e_src, e_dst, dinv, bufB, E);
    ln_relu_k<<<cdiv_ll(N, 8), blk, 0, stream>>>(bufB, ln1w, ln1b, N);
    // ---- GCN layer 2
    run_gemm(bufB, W_g2, nullptr, bufA, 128, 0);
    gcn_self_init_k<<<cdiv_ll(N * 32, 256), blk, 0, stream>>>(bufA, dinv, b_g2, bufB, N);
    gcn_scatter_k<<<cdiv_ll(E * 32, 256), blk, 0, stream>>>(bufA, e_src, e_dst, dinv, bufB, E);
    ln_relu_k<<<cdiv_ll(N, 8), blk, 0, stream>>>(bufB, ln1w, ln1b, N);
    // bufB = h1

    // ---- HGC branch: layer 1
    run_gemm(x, W_h1, nullptr, bufA, 512, 0);
    fill_k<<<cdiv_ll(NHE * 128, 256), blk, 0, stream>>>(bufM, 0.0f, NHE * 128);
    scatter_rows_k<<<cdiv_ll(HM * 32, 256), blk, 0, stream>>>(bufA, h_node, h_hedge, bufM, HM);
    scale_rows_k<<<cdiv_ll(NHE * 32, 256), blk, 0, stream>>>(bufM, Binv, NHE);
    fill_k<<<cdiv_ll(N * 128, 256), blk, 0, stream>>>(bufC, 0.0f, N * 128);
    scatter_rows_k<<<cdiv_ll(HM * 32, 256), blk, 0, stream>>>(bufM, h_hedge, h_node, bufC, HM);
    scale_rows_bias_k<<<cdiv_ll(N * 32, 256), blk, 0, stream>>>(bufC, Dinv, b_h1, N);
    ln_relu_k<<<cdiv_ll(N, 8), blk, 0, stream>>>(bufC, ln2w, ln2b, N);
    // ---- HGC layer 2
    run_gemm(bufC, W_h2, nullptr, bufA, 128, 0);
    fill_k<<<cdiv_ll(NHE * 128, 256), blk, 0, stream>>>(bufM, 0.0f, NHE * 128);
    scatter_rows_k<<<cdiv_ll(HM * 32, 256), blk, 0, stream>>>(bufA, h_node, h_hedge, bufM, HM);
    scale_rows_k<<<cdiv_ll(NHE * 32, 256), blk, 0, stream>>>(bufM, Binv, NHE);
    fill_k<<<cdiv_ll(N * 128, 256), blk, 0, stream>>>(bufC, 0.0f, N * 128);
    scatter_rows_k<<<cdiv_ll(HM * 32, 256), blk, 0, stream>>>(bufM, h_hedge, h_node, bufC, HM);
    scale_rows_bias_k<<<cdiv_ll(N * 32, 256), blk, 0, stream>>>(bufC, Dinv, b_h2, N);
    ln_relu_k<<<cdiv_ll(N, 8), blk, 0, stream>>>(bufC, ln2w, ln2b, N);
    // bufC = h2

    // ---- combine + MLP head
    avg_k<<<cdiv_ll(N * 32, 256), blk, 0, stream>>>(bufB, bufC, bufA, N * 32);
    run_gemm(bufA, Wp1, bp1, bufB, 128, 1);
    run_gemm(bufB, Wp2, bp2, (float*)d_out, 128, 0);
}
